// FIBERTransformerBlock_53317724013253
// MI455X (gfx1250) — matrix operand order repaired and checked by reading, not compiled, not run on hardware
//
#include <hip/hip_runtime.h>
#include <hip/hip_bf16.h>
#include <math.h>

typedef __bf16 bf16_t;
typedef __attribute__((ext_vector_type(16))) __bf16 v16bf;
typedef __attribute__((ext_vector_type(8)))  __bf16 v8bf;
typedef __attribute__((ext_vector_type(8)))  float  v8f;
typedef __attribute__((ext_vector_type(2)))  unsigned int u32x2;

#define WMMA_BF16(a, b, c) \
  __builtin_amdgcn_wmma_f32_16x16x32_bf16(false, (a), false, (b), (short)0, (c), false, false)

__device__ __forceinline__ bf16_t f2bf(float f) {
  union { float f; unsigned int u; } a; a.f = f;
  unsigned int u = a.u + 0x7FFFu + ((a.u >> 16) & 1u);   // round-to-nearest-even
  union { unsigned short s; bf16_t b; } r; r.s = (unsigned short)(u >> 16);
  return r.b;
}
// pack two f32 -> two rounded bf16 in one dword (elem a = low half, b = high half)
__device__ __forceinline__ unsigned int pk2(float a, float b) {
  union { float f; unsigned int u; } x, y; x.f = a; y.f = b;
  unsigned int ua = x.u + 0x7FFFu + ((x.u >> 16) & 1u);
  unsigned int ub = y.u + 0x7FFFu + ((y.u >> 16) & 1u);
  return (ua >> 16) | (ub & 0xFFFF0000u);
}
__device__ __forceinline__ void st4bf(bf16_t* p, float a, float b, float c, float d) {
  u32x2 v; v.x = pk2(a, b); v.y = pk2(c, d);
  *(u32x2*)p = v;                                        // one ds_store_b64
}

// A-fragment (16x32 bf16): lane ll = row, lane-half lh picks K={kh*8..+7, 16+kh*8..+7}
__device__ __forceinline__ v16bf load_a_frag(const bf16_t* base, int stride, int row, int kOff) {
  union { v16bf v; v8bf h[2]; } u;
  const bf16_t* p = base + row * stride + kOff;
  u.h[0] = *(const v8bf*)(p);
  u.h[1] = *(const v8bf*)(p + 16);
  return u.v;
}
// B-fragment (32x16 bf16) from [n][k] LDS: lane-half lh picks K={kh*8..+7, 16+kh*8..+7}
__device__ __forceinline__ v16bf load_b_frag(const bf16_t* base, int stride, int col, int kOff) {
  union { v16bf v; v8bf h[2]; } u;
  const bf16_t* p = base + col * stride + kOff;
  u.h[0] = *(const v8bf*)(p);
  u.h[1] = *(const v8bf*)(p + 16);
  return u.v;
}

// ---------------- GEMM: C[M,N] = act(A[M,K] @ W[K,N] + bias (+ res)) ----------------
#define LDS_STR 48   // 32 K-elems + pad; row pitch 96B keeps 16B-aligned b128 frag reads

__global__ __launch_bounds__(256) void gemm_kernel(
    const float* __restrict__ A, const float* __restrict__ W,
    const float* __restrict__ bias, const float* __restrict__ res,
    float* __restrict__ C, int M, int N, int K, int act)
{
  __shared__ alignas(16) bf16_t As[128 * LDS_STR];
  __shared__ alignas(16) bf16_t Bs[128 * LDS_STR];

  const int tid  = threadIdx.x;
  const int lane = tid & 31;
  const int wave = tid >> 5;
  const int wm = wave >> 1;   // 0..3 -> 32-row subtile
  const int wn = wave & 1;    // 0..1 -> 64-col subtile
  const int lh = lane >> 4, ll = lane & 15;

  const int n0 = blockIdx.x * 128;
  const int m0 = blockIdx.y * 128;

  v8f c[2][4] = {};

  for (int k0 = 0; k0 < K; k0 += 32) {
    // prefetch next K-tile (global_prefetch_b8)
    if (k0 + 32 < K) {
      __builtin_prefetch(A + (size_t)(m0 + (tid >> 1)) * K + k0 + 32, 0, 1);
      __builtin_prefetch(W + (size_t)(k0 + 32 + (tid >> 3)) * N + n0 + (tid & 7) * 16, 0, 1);
    }
    // stage A tile 128x32 (k-contiguous): float4 load -> packed b64 store
    #pragma unroll
    for (int i = 0; i < 4; i++) {
      int e = (i * 256 + tid) * 4;
      int row = e >> 5, k = e & 31;
      const float4 f = *(const float4*)(A + (size_t)(m0 + row) * K + k0 + k);
      st4bf(As + row * LDS_STR + k, f.x, f.y, f.z, f.w);
    }
    // stage W tile 32x128 transposed -> Bs[n][k]: column-of-4 loads, packed b64 store
    #pragma unroll
    for (int i = 0; i < 4; i++) {
      int g = i * 256 + tid;           // 0..1023
      int n = g & 127, kg = g >> 7;    // kg in 0..7 -> k = kg*4
      const float* wp = W + (size_t)(k0 + kg * 4) * N + n0 + n;
      float w0 = wp[0];
      float w1 = wp[(size_t)N];
      float w2 = wp[(size_t)2 * N];
      float w3 = wp[(size_t)3 * N];
      st4bf(Bs + n * LDS_STR + kg * 4, w0, w1, w2, w3);
    }
    __syncthreads();

    v16bf a0 = load_a_frag(As + (wm * 32 +  0) * LDS_STR, LDS_STR, ll, lh * 8);
    v16bf a1 = load_a_frag(As + (wm * 32 + 16) * LDS_STR, LDS_STR, ll, lh * 8);
    #pragma unroll
    for (int ni = 0; ni < 4; ni++) {
      v16bf b = load_b_frag(Bs + (wn * 64 + ni * 16) * LDS_STR, LDS_STR, ll, lh * 8);
      c[0][ni] = WMMA_BF16(a0, b, c[0][ni]);
      c[1][ni] = WMMA_BF16(a1, b, c[1][ni]);
    }
    __syncthreads();
  }

  #pragma unroll
  for (int mi = 0; mi < 2; mi++) {
    #pragma unroll
    for (int ni = 0; ni < 4; ni++) {
      #pragma unroll
      for (int r = 0; r < 8; r++) {
        int row = m0 + wm * 32 + mi * 16 + r + 8 * lh;
        int col = n0 + wn * 64 + ni * 16 + ll;
        float v = c[mi][ni][r] + bias[col];
        if (res) v += res[(size_t)row * N + col];
        if (act == 1) v = 0.5f * v * (1.0f + erff(v * 0.70710678118654752f)); // exact GELU
        C[(size_t)row * N + col] = v;
      }
    }
  }
}

// ---------------- Flash-style attention: O = softmax(Q K^T * scale) V ----------------
// grid = (Sq/64, H, B), block = 128 (4 waves, 16 q-rows each). dh = 64.
__global__ __launch_bounds__(128) void attn_kernel(
    const float* __restrict__ Q, const float* __restrict__ K,
    const float* __restrict__ V, float* __restrict__ O,
    int Sq, int Skv, int qStride, int kvStride, float scale)
{
  __shared__ alignas(16) bf16_t Qs[64 * 64];     // [qrow][dh]
  __shared__ alignas(16) bf16_t Ks[32 * 64];     // [key][dh]
  __shared__ alignas(16) bf16_t Vt[64 * 32];     // [dh][key]  (transposed)
  __shared__ alignas(16) bf16_t Ps[4][16 * 32];  // per-wave P tile [qrow][key]

  const int tid = threadIdx.x;
  const int lane = tid & 31;
  const int w = tid >> 5;
  const int lh = lane >> 4, ll = lane & 15;
  const int b = blockIdx.z, h = blockIdx.y;
  const int q0 = blockIdx.x * 64;
  const int D = 1024;

  // stage Q tile 64x64 (dh contiguous): float4 load -> packed b64 store
  #pragma unroll
  for (int i = 0; i < 8; i++) {
    int e = (i * 128 + tid) * 4;
    int row = e >> 6, d = e & 63;
    const float4 f = *(const float4*)(Q + ((size_t)b * Sq + q0 + row) * qStride + h * 64 + d);
    st4bf(Qs + row * 64 + d, f.x, f.y, f.z, f.w);
  }

  v8f ctx[4] = {};
  float mo[8], ls[8];
  #pragma unroll
  for (int r = 0; r < 8; r++) { mo[r] = -3.0e38f; ls[r] = 0.0f; }

  const int nb = Skv / 32;
  for (int kb = 0; kb < nb; kb++) {
    __syncthreads();   // protects Ks/Vt reuse across iterations (and publishes Qs on iter 0)
    // stage K block 32x64 (natural layout)
    #pragma unroll
    for (int i = 0; i < 4; i++) {
      int e = (i * 128 + tid) * 4;
      int kr = e >> 6, d = e & 63;
      const float4 fk = *(const float4*)(K + ((size_t)b * Skv + kb * 32 + kr) * kvStride + h * 64 + d);
      st4bf(Ks + kr * 64 + d, fk.x, fk.y, fk.z, fk.w);
    }
    // stage V block transposed -> Vt[dh][key]: column-of-4 loads, packed b64 store
    #pragma unroll
    for (int i = 0; i < 4; i++) {
      int g = i * 128 + tid;          // 0..511
      int d = g & 63, kg = g >> 6;    // kg in 0..7 -> key = kg*4
      const float* vp = V + ((size_t)b * Skv + kb * 32 + kg * 4) * kvStride + h * 64 + d;
      float v0 = vp[0];
      float v1 = vp[(size_t)kvStride];
      float v2 = vp[(size_t)2 * kvStride];
      float v3 = vp[(size_t)3 * kvStride];
      st4bf(Vt + d * 32 + kg * 4, v0, v1, v2, v3);
    }
    __syncthreads();

    // scores: (16 q-rows) x (32 keys), accumulate over dh=64 in two K=32 steps
    v8f s0 = {}, s1 = {};
    #pragma unroll
    for (int ks = 0; ks < 2; ks++) {
      v16bf aq = load_a_frag(Qs + (w * 16) * 64, 64, ll, ks * 32 + lh * 8);
      v16bf b0 = load_b_frag(Ks,            64, ll, ks * 32 + lh * 8);
      v16bf b1 = load_b_frag(Ks + 16 * 64,  64, ll, ks * 32 + lh * 8);
      s0 = WMMA_BF16(aq, b0, s0);
      s1 = WMMA_BF16(aq, b1, s1);
    }

    // streaming softmax per q-row (row r lives in VGPR r, striped across a 16-lane half)
    bf16_t* Pw = &Ps[w][0];
    #pragma unroll
    for (int r = 0; r < 8; r++) {
      float x0 = s0[r] * scale, x1 = s1[r] * scale;
      float t = fmaxf(x0, x1);
      t = fmaxf(t, __shfl_xor(t, 1, 32));
      t = fmaxf(t, __shfl_xor(t, 2, 32));
      t = fmaxf(t, __shfl_xor(t, 4, 32));
      t = fmaxf(t, __shfl_xor(t, 8, 32));          // masks < 16: stays within half
      float mn = fmaxf(mo[r], t);
      float alpha = __expf(mo[r] - mn);
      mo[r] = mn;
      float p0 = __expf(x0 - mn), p1 = __expf(x1 - mn);
      float rs = p0 + p1;
      rs += __shfl_xor(rs, 1, 32);
      rs += __shfl_xor(rs, 2, 32);
      rs += __shfl_xor(rs, 4, 32);
      rs += __shfl_xor(rs, 8, 32);
      ls[r] = ls[r] * alpha + rs;
      #pragma unroll
      for (int ni = 0; ni < 4; ni++) ctx[ni][r] *= alpha;
      Pw[(r + 8 * lh) * 32 + ll]      = f2bf(p0);  // C-layout -> A-layout via LDS
      Pw[(r + 8 * lh) * 32 + 16 + ll] = f2bf(p1);
    }

    // ctx += P (16x32) @ V (32x64)
    v16bf ap = load_a_frag(Pw, 32, ll, lh * 8);
    #pragma unroll
    for (int ni = 0; ni < 4; ni++) {
      v16bf bv = load_b_frag(Vt + (ni * 16) * 32, 32, ll, lh * 8);
      ctx[ni] = WMMA_BF16(ap, bv, ctx[ni]);
    }
  }

  #pragma unroll
  for (int ni = 0; ni < 4; ni++) {
    #pragma unroll
    for (int r = 0; r < 8; r++) {
      int row = q0 + w * 16 + r + 8 * lh;
      int col = h * 64 + ni * 16 + ll;
      O[((size_t)b * Sq + row) * D + col] = ctx[ni][r] / ls[r];
    }
  }
}

// ---------------- LayerNorm: one block per row, D = 1024 ----------------
__global__ __launch_bounds__(256) void ln_kernel(
    const float* __restrict__ x, const float* __restrict__ s,
    const float* __restrict__ b, float* __restrict__ y)
{
  __shared__ float rs[256], rq[256];
  const int row = blockIdx.x, tid = threadIdx.x;
  const int D = 1024;
  const float* xr = x + (size_t)row * D;
  float sm = 0.f, sq = 0.f, vals[4];
  #pragma unroll
  for (int j = 0; j < 4; j++) { float v = xr[tid + j * 256]; vals[j] = v; sm += v; sq += v * v; }
  rs[tid] = sm; rq[tid] = sq; __syncthreads();
  for (int off = 128; off > 0; off >>= 1) {
    if (tid < off) { rs[tid] += rs[tid + off]; rq[tid] += rq[tid + off]; }
    __syncthreads();
  }
  float mean = rs[0] * (1.0f / 1024.0f);
  float var  = rq[0] * (1.0f / 1024.0f) - mean * mean;
  float rstd = rsqrtf(var + 1e-12f);
  float* yr = y + (size_t)row * D;
  #pragma unroll
  for (int j = 0; j < 4; j++) { int d = tid + j * 256; yr[d] = (vals[j] - mean) * rstd * s[d] + b[d]; }
}

// ---------------- Host orchestration ----------------
extern "C" void kernel_launch(void* const* d_in, const int* in_sizes, int n_in,
                              void* d_out, int out_size, void* d_ws, size_t ws_size,
                              hipStream_t stream) {
  (void)in_sizes; (void)n_in; (void)out_size; (void)ws_size;
  const int B = 8, Sv = 576, St = 512, D = 1024, I = 4096, H = 16;
  const int Mv = B * Sv, Mt = B * St;
  const size_t NVf = (size_t)Mv * D, NTf = (size_t)Mt * D;
  const float scale = 0.125f; // 1/sqrt(64)

  auto F = [&](int i) { return (const float*)d_in[i]; };
  const float* vis_in = F(0);
  const float* txt_in = F(1);
  // v: 2..13  t: 14..25  (ln1_s, ln1_b, ln2_s, ln2_b, in_w, in_b, out_w, out_b, f1w, f1b, f2w, f2b)
  // cm: vq 26/27, tk 28/29, tv 30/31, tq 32/33, vk 34/35, vv 36/37, vout 38/39, tout 40/41
  // cm_vln 42/43, cm_tln 44/45

  float* ws  = (float*)d_ws;
  float* VIS = ws;               // current vis activations
  float* TXT = VIS + NVf;        // current txt activations
  float* T0  = TXT + NTf;        // tmp (size NVf)
  float* T1  = T0 + NVf;         // tmp (size NVf)
  float* BIG = T1 + NVf;         // size Mv*I = 18.9M floats (qkv / cross-qkv / ffn hidden)

  float* OUTv = (float*)d_out;
  float* OUTt = OUTv + NVf;

  auto gemm = [&](const float* A, const float* W, const float* bias, const float* res,
                  float* C, int M, int N, int K, int act) {
    dim3 g(N / 128, M / 128);
    gemm_kernel<<<g, 256, 0, stream>>>(A, W, bias, res, C, M, N, K, act);
  };
  auto lnk = [&](const float* x, const float* s, const float* b, float* y, int rows) {
    ln_kernel<<<rows, 256, 0, stream>>>(x, s, b, y);
  };
  auto attn = [&](const float* Q, const float* K, const float* V, float* O,
                  int Sq, int Skv, int qStr, int kvStr) {
    dim3 g(Sq / 64, H, B);
    attn_kernel<<<g, 128, 0, stream>>>(Q, K, V, O, Sq, Skv, qStr, kvStr, scale);
  };

  // ---- vis self-attention + residual ----
  lnk(vis_in, F(2), F(3), T0, Mv);
  gemm(T0, F(6), F(7), nullptr, BIG, Mv, 3 * D, D, 0);                 // qkv
  attn(BIG, BIG + 1024, BIG + 2048, T1, Sv, Sv, 3 * D, 3 * D);
  gemm(T1, F(8), F(9), vis_in, VIS, Mv, D, D, 0);                      // out-proj + residual

  // ---- txt self-attention + residual ----
  lnk(txt_in, F(14), F(15), T0, Mt);
  gemm(T0, F(18), F(19), nullptr, BIG, Mt, 3 * D, D, 0);
  attn(BIG, BIG + 1024, BIG + 2048, T1, St, St, 3 * D, 3 * D);
  gemm(T1, F(20), F(21), txt_in, TXT, Mt, D, D, 0);

  // ---- cross-modal: v2t = attend(vq, tk, tv) ----
  {
    float* CQ = BIG; float* CK = CQ + NVf; float* CV = CK + NTf;
    gemm(VIS, F(26), F(27), nullptr, CQ, Mv, D, D, 0);
    gemm(TXT, F(28), F(29), nullptr, CK, Mt, D, D, 0);
    gemm(TXT, F(30), F(31), nullptr, CV, Mt, D, D, 0);
    attn(CQ, CK, CV, T1, Sv, St, D, D);                                // v2t -> T1
  }
  // ---- cross-modal: t2v = attend(tq, vk, vv) ----
  {
    float* CQ = BIG; float* CK = CQ + NTf; float* CV = CK + NVf;
    gemm(TXT, F(32), F(33), nullptr, CQ, Mt, D, D, 0);
    gemm(VIS, F(34), F(35), nullptr, CK, Mv, D, D, 0);
    gemm(VIS, F(36), F(37), nullptr, CV, Mv, D, D, 0);
    attn(CQ, CK, CV, T0, St, Sv, D, D);                                // t2v -> T0
  }
  // out-proj + residual, then LN
  gemm(T1, F(38), F(39), VIS, BIG, Mv, D, D, 0);
  lnk(BIG, F(42), F(43), VIS, Mv);
  gemm(T0, F(40), F(41), TXT, BIG, Mt, D, D, 0);
  lnk(BIG, F(44), F(45), TXT, Mt);

  // ---- vis FFN + residual ----
  lnk(VIS, F(4), F(5), T0, Mv);
  gemm(T0, F(10), F(11), nullptr, BIG, Mv, I, D, 1);                   // GELU
  gemm(BIG, F(12), F(13), VIS, OUTv, Mv, D, I, 0);

  // ---- txt FFN + residual ----
  lnk(TXT, F(16), F(17), T0, Mt);
  gemm(T0, F(22), F(23), nullptr, BIG, Mt, I, D, 1);
  gemm(BIG, F(24), F(25), TXT, OUTt, Mt, D, I, 0);
}
